// WindowAttentionBlock_79078937854089
// MI455X (gfx1250) — matrix operand order repaired and checked by reading, not compiled, not run on hardware
//
#include <hip/hip_runtime.h>
#include <hip/hip_bf16.h>
#include <math.h>

// ---------------------------------------------------------------------------
// CDNA5 (gfx1250, wave32) Swin window-attention block, bf16 WMMA everywhere.
// ~290 GFLOP total; all 4 big GEMMs + attention use v_wmma_f32_16x16x32_bf16.
// Workspace requirement ~612 MB (packed with lifetime reuse).
// ---------------------------------------------------------------------------

typedef __bf16 bf16;
typedef bf16  v16bf __attribute__((ext_vector_type(16)));
typedef float v8f   __attribute__((ext_vector_type(8)));
typedef float f4    __attribute__((ext_vector_type(4)));

#define DIM      384
#define HEADS    12
#define HD       32
#define NTOK     49          // 7*7 tokens per window
#define TOKPAD   64          // padded to 64 rows for 16-row WMMA tiles
#define HID      1024
#define BATCH    32
#define HW       56
#define SPAT     (56*56)     // 3136
#define NWIN     (32*8*8)    // 2048 windows
#define MWIN     ((size_t)NWIN*TOKPAD)   // 131072 padded token-rows
#define NTOKENS  ((size_t)BATCH*SPAT)    // 100352 real tokens (=784*128)

union FA { v16bf v; f4 q[2]; bf16 s[16]; };
union FC { v8f  v; float s[8]; };

// A fragment: 16x32 bf16, row-major A (lda = K). Lane l: row m=l&15, kh=l>>4.
// elems 0..7  -> K = kh*8 + e        (contiguous 16B)
// elems 8..15 -> K = 16 + kh*8 + e-8 (contiguous 16B)
__device__ __forceinline__ FA load_frag_a(const bf16* __restrict__ A, int lda) {
  const int lane = threadIdx.x & 31;
  const int m = lane & 15, kh = lane >> 4;
  const bf16* p = A + (size_t)m * lda + kh * 8;
  FA f;
  f.q[0] = *(const f4*)p;
  f.q[1] = *(const f4*)(p + 16);
  return f;
}

// B fragment: 32x16 bf16, B given as W[N][K] row-major (computes A @ W^T).
// Lane l: col n=l&15 (row n of W), kh=l>>4; elems 0..7 -> K = kh*8 + e, elems 8..15 -> K = 16 + kh*8 + e-8.
__device__ __forceinline__ FA load_frag_b(const bf16* __restrict__ W, int ldb) {
  const int lane = threadIdx.x & 31;
  const int n = lane & 15, kh = lane >> 4;
  const bf16* p = W + (size_t)n * ldb + kh * 8;
  FA f;
  f.q[0] = *(const f4*)p;
  f.q[1] = *(const f4*)(p + 16);
  return f;
}

__device__ __forceinline__ v8f wmma_bf16(const FA& a, const FA& b, v8f c) {
  return __builtin_amdgcn_wmma_f32_16x16x32_bf16(
      false, a.v, false, b.v, (short)0, c, false, false);
}

// ---------------------------------------------------------------------------
// Generic bf16 GEMM: out = A[M,K] @ W[N,K]^T (+epilogue)
// Block: 256 thr = 8 waves -> 128x128 tile. Wave: 64(M)x32(N) = 4x2 WMMA tiles.
// EPI 0: QKV   -> bf16 out (ldo), +bias
// EPI 1: PROJ  -> un-window rows, +bias, +x residual (NCHW), fp32 x2[t][c]
// EPI 2: FC1   -> bf16 out (ldo), +bias, exact GELU
// EPI 3: FC2   -> +bias, +x2 residual, fp32 NCHW scatter store
// ---------------------------------------------------------------------------
template <int EPI>
__global__ __launch_bounds__(256) void gemm_bf16_kernel(
    const bf16* __restrict__ A, const bf16* __restrict__ W,
    const float* __restrict__ bias, int K, int ldo,
    bf16* __restrict__ outb, float* __restrict__ outf,
    const float* __restrict__ resid) {
  const int lane = threadIdx.x & 31;
  const int wave = threadIdx.x >> 5;
  const int mw = wave & 1;      // 2 M-waves of 64
  const int nw = wave >> 1;     // 4 N-waves of 32
  const int m0 = blockIdx.y * 128 + mw * 64;
  const int n0 = blockIdx.x * 128 + nw * 32;
  const int nl = lane & 15, kh = lane >> 4;

  FC acc[4][2];
#pragma unroll
  for (int i = 0; i < 4; i++)
#pragma unroll
    for (int j = 0; j < 2; j++) {
      v8f z = {0.f, 0.f, 0.f, 0.f, 0.f, 0.f, 0.f, 0.f};
      acc[i][j].v = z;
    }

  for (int k0 = 0; k0 < K; k0 += 32) {
    FA fb0 = load_frag_b(W + (size_t)n0 * K + k0, K);
    FA fb1 = load_frag_b(W + (size_t)(n0 + 16) * K + k0, K);
#pragma unroll
    for (int mt = 0; mt < 4; mt++) {
      FA fa = load_frag_a(A + (size_t)(m0 + mt * 16) * K + k0, K);
      acc[mt][0].v = wmma_bf16(fa, fb0, acc[mt][0].v);
      acc[mt][1].v = wmma_bf16(fa, fb1, acc[mt][1].v);
    }
  }

  // Epilogue. C layout: col n = n0+nt*16+nl ; row m = m0+mt*16+r+8*kh.
#pragma unroll
  for (int nt = 0; nt < 2; nt++) {
    const int n = n0 + nt * 16 + nl;
    const float bn = bias[n];
#pragma unroll
    for (int mt = 0; mt < 4; mt++) {
#pragma unroll
      for (int r = 0; r < 8; r++) {
        const int m = m0 + mt * 16 + r + 8 * kh;
        const float v = acc[mt][nt].s[r] + bn;
        if (EPI == 0) {  // QKV: bf16 out
          outb[(size_t)m * ldo + n] = (bf16)v;
        } else if (EPI == 1) {  // PROJ: un-window + residual -> x2 fp32
          const int win = m >> 6, tok = m & 63;
          if (tok < NTOK) {
            const int b = win >> 6, wh = (win >> 3) & 7, ww = win & 7;
            const int hh = wh * 7 + tok / 7, wl = ww * 7 + tok % 7;
            const int s = hh * HW + wl;
            const size_t t = (size_t)b * SPAT + s;
            const float rsd = resid[((size_t)b * DIM + n) * SPAT + s];
            outf[t * DIM + n] = v + rsd;
          }
        } else if (EPI == 2) {  // FC1 + exact GELU -> bf16
          const float g = 0.5f * v * (1.f + erff(v * 0.70710678118654752f));
          outb[(size_t)m * ldo + n] = (bf16)g;
        } else {  // FC2 + x2 residual -> fp32 NCHW
          const int b = m / SPAT, s = m % SPAT;
          outf[((size_t)b * DIM + n) * SPAT + s] = v + resid[(size_t)m * DIM + n];
        }
      }
    }
  }
}

// ---------------------------------------------------------------------------
// LN1 + window partition: x [B,C,56,56] fp32 -> a1 [2048,64,384] bf16.
// Block handles (b, h, half-row of 28 w): coalesced along W, LDS transpose.
// ---------------------------------------------------------------------------
__global__ __launch_bounds__(256) void ln1_window_kernel(
    const float* __restrict__ x, const float* __restrict__ g,
    const float* __restrict__ bta, bf16* __restrict__ a1) {
  __shared__ float tile[DIM * 29];  // [c][wl] pad 29, 44.5 KB
  __shared__ float smu[28], srs[28];
  const int bh = blockIdx.x;             // 0 .. 32*56*2-1
  const int b = bh / 112, rem = bh % 112;
  const int h = rem >> 1, w0 = (rem & 1) * 28;
  const float* xb = x + ((size_t)b * DIM) * SPAT + h * HW + w0;

  for (int i = threadIdx.x; i < DIM * 28; i += 256) {
    const int c = i / 28, wl = i % 28;
    tile[c * 29 + wl] = xb[(size_t)c * SPAT + wl];
  }
  __syncthreads();
  if (threadIdx.x < 28) {
    const int wl = threadIdx.x;
    float s = 0.f, s2 = 0.f;
    for (int c = 0; c < DIM; c++) {
      const float v = tile[c * 29 + wl];
      s += v;
      s2 += v * v;
    }
    const float mu = s * (1.f / DIM);
    const float var = s2 * (1.f / DIM) - mu * mu;
    smu[wl] = mu;
    srs[wl] = rsqrtf(var + 1e-5f);
  }
  __syncthreads();
  for (int i = threadIdx.x; i < DIM * 28; i += 256) {
    const int wl = i / DIM, c = i % DIM;
    const int w = w0 + wl;
    const float v = (tile[c * 29 + wl] - smu[wl]) * srs[wl] * g[c] + bta[c];
    const int win = b * 64 + (h / 7) * 8 + (w / 7);
    const int tok = (h % 7) * 7 + (w % 7);
    a1[((size_t)win * TOKPAD + tok) * DIM + c] = (bf16)v;
  }
}

// Zero the padded token rows (49..63) so padded Q/K/V stay benign.
__global__ void zero_pad_kernel(bf16* __restrict__ a1) {
  const size_t idx = (size_t)blockIdx.x * 256 + threadIdx.x;
  const size_t total = (size_t)NWIN * 15 * DIM;
  if (idx >= total) return;
  const int win = (int)(idx / (15 * DIM));
  const int rem = (int)(idx % (15 * DIM));
  const int tok = 49 + rem / DIM, c = rem % DIM;
  a1[((size_t)win * TOKPAD + tok) * DIM + c] = (bf16)0.f;
}

__global__ void cvt_f32_bf16_kernel(const float* __restrict__ s,
                                    bf16* __restrict__ d, int n) {
  const int i = blockIdx.x * 256 + threadIdx.x;
  if (i < n) d[i] = (bf16)s[i];
}

// ---------------------------------------------------------------------------
// Attention: one wave per (window, head). 2 waves/block (2 heads).
// QKV layout [win][tok64][1152]; head slices of 32 contiguous channels.
// S = Q K^T (16 WMMAs), masked softmax over cols<49, O = P V (16 WMMAs).
// ---------------------------------------------------------------------------
__global__ __launch_bounds__(64) void attn_kernel(
    const bf16* __restrict__ qkv, bf16* __restrict__ o) {
  const int lane = threadIdx.x & 31;
  const int wsub = threadIdx.x >> 5;  // 0..1
  const int win = blockIdx.x;
  const int head = blockIdx.y * 2 + wsub;
  const int nl = lane & 15, kh = lane >> 4;
  __shared__ float Pl[2][64 * 65];  // 33.3 KB
  float* P = Pl[wsub];

  const int ld = 3 * DIM;  // 1152
  const bf16* base = qkv + (size_t)win * TOKPAD * ld;
  const bf16* Q = base + head * HD;
  const bf16* Km = base + DIM + head * HD;
  const bf16* V = base + 2 * DIM + head * HD;

  FA fq[4], fk[4];
#pragma unroll
  for (int mt = 0; mt < 4; mt++) fq[mt] = load_frag_a(Q + (size_t)mt * 16 * ld, ld);
#pragma unroll
  for (int nt = 0; nt < 4; nt++) fk[nt] = load_frag_b(Km + (size_t)nt * 16 * ld, ld);

  FC S[4][4];
#pragma unroll
  for (int mt = 0; mt < 4; mt++)
#pragma unroll
    for (int nt = 0; nt < 4; nt++) {
      v8f z = {0.f, 0.f, 0.f, 0.f, 0.f, 0.f, 0.f, 0.f};
      S[mt][nt].v = wmma_bf16(fq[mt], fk[nt], z);
    }

  const float scale = 0.17677669529663687f;  // 1/sqrt(32)
#pragma unroll
  for (int mt = 0; mt < 4; mt++) {
#pragma unroll
    for (int r = 0; r < 8; r++) {
      float vals[4];
      float mx = -INFINITY;
#pragma unroll
      for (int nt = 0; nt < 4; nt++) {
        const int n = nt * 16 + nl;
        const float v = (n < NTOK) ? S[mt][nt].s[r] * scale : -INFINITY;
        vals[nt] = v;
        mx = fmaxf(mx, v);
      }
#pragma unroll
      for (int off = 8; off > 0; off >>= 1) mx = fmaxf(mx, __shfl_xor(mx, off, 32));
      float sum = 0.f, e[4];
#pragma unroll
      for (int nt = 0; nt < 4; nt++) {
        e[nt] = (nt * 16 + nl < NTOK) ? __expf(vals[nt] - mx) : 0.f;
        sum += e[nt];
      }
#pragma unroll
      for (int off = 8; off > 0; off >>= 1) sum += __shfl_xor(sum, off, 32);
      const float inv = 1.f / sum;
      const int m = mt * 16 + r + 8 * kh;
#pragma unroll
      for (int nt = 0; nt < 4; nt++) P[m * 65 + nt * 16 + nl] = e[nt] * inv;
    }
  }
  __syncthreads();

  FC O[4][2];
#pragma unroll
  for (int mt = 0; mt < 4; mt++)
#pragma unroll
    for (int nt = 0; nt < 2; nt++) {
      v8f z = {0.f, 0.f, 0.f, 0.f, 0.f, 0.f, 0.f, 0.f};
      O[mt][nt].v = z;
    }

#pragma unroll
  for (int ks = 0; ks < 2; ks++) {
    FA fv[2];
#pragma unroll
    for (int nt = 0; nt < 2; nt++) {
      const int n = nt * 16 + nl;  // hd column
#pragma unroll
      for (int e = 0; e < 16; e++)
        fv[nt].s[e] = V[(size_t)(ks * 32 + ((e < 8) ? kh * 8 + e : 16 + kh * 8 + (e - 8))) * ld + n];
    }
#pragma unroll
    for (int mt = 0; mt < 4; mt++) {
      FA fp;  // P fragment, A layout, converted fp32 -> bf16
      const int row = mt * 16 + nl;
#pragma unroll
      for (int e = 0; e < 16; e++) {
        const int k = ks * 32 + ((e < 8) ? kh * 8 + e : 16 + kh * 8 + (e - 8));
        fp.s[e] = (bf16)P[row * 65 + k];
      }
      O[mt][0].v = wmma_bf16(fp, fv[0], O[mt][0].v);
      O[mt][1].v = wmma_bf16(fp, fv[1], O[mt][1].v);
    }
  }

#pragma unroll
  for (int mt = 0; mt < 4; mt++)
#pragma unroll
    for (int nt = 0; nt < 2; nt++)
#pragma unroll
      for (int r = 0; r < 8; r++) {
        const int m = mt * 16 + r + 8 * kh;
        const int n = nt * 16 + nl;
        o[((size_t)win * TOKPAD + m) * DIM + head * HD + n] = (bf16)O[mt][nt].s[r];
      }
}

// LN2: x2 [tokens,384] fp32 -> a2 bf16. One block (384 thr = 12 waves) / token.
__global__ __launch_bounds__(384) void ln2_kernel(
    const float* __restrict__ x2, const float* __restrict__ g,
    const float* __restrict__ bta, bf16* __restrict__ a2) {
  __shared__ float red[DIM], red2[DIM];
  const size_t t = blockIdx.x;
  const int c = threadIdx.x;
  const float v = x2[t * DIM + c];
  red[c] = v;
  red2[c] = v * v;
  __syncthreads();
  for (int s = 256; s > 0; s >>= 1) {
    if (c < s && c + s < DIM) {
      red[c] += red[c + s];
      red2[c] += red2[c + s];
    }
    __syncthreads();
  }
  const float mu = red[0] * (1.f / DIM);
  const float var = red2[0] * (1.f / DIM) - mu * mu;
  const float rs = rsqrtf(var + 1e-5f);
  a2[t * DIM + c] = (bf16)((v - mu) * rs * g[c] + bta[c]);
}

// ---------------------------------------------------------------------------
extern "C" void kernel_launch(void* const* d_in, const int* in_sizes, int n_in,
                              void* d_out, int out_size, void* d_ws,
                              size_t ws_size, hipStream_t stream) {
  (void)in_sizes; (void)n_in; (void)out_size; (void)ws_size;
  const float* x      = (const float*)d_in[0];
  const float* ln1_g  = (const float*)d_in[1];
  const float* ln1_b  = (const float*)d_in[2];
  const float* qkv_w  = (const float*)d_in[3];
  const float* qkv_b  = (const float*)d_in[4];
  const float* proj_w = (const float*)d_in[5];
  const float* proj_b = (const float*)d_in[6];
  const float* ln2_g  = (const float*)d_in[7];
  const float* ln2_b  = (const float*)d_in[8];
  const float* fc1_w  = (const float*)d_in[9];
  const float* fc1_b  = (const float*)d_in[10];
  const float* fc2_w  = (const float*)d_in[11];
  const float* fc2_b  = (const float*)d_in[12];
  float* out = (float*)d_out;

  // Workspace packing (peak ~612 MB), all offsets 256B aligned.
  char* ws = (char*)d_ws;
  size_t off = 0;
  auto alloc = [&](size_t bytes) {
    char* p = ws + off;
    off += (bytes + 255) & ~(size_t)255;
    return p;
  };
  bf16* w_qkv  = (bf16*)alloc((size_t)3 * DIM * DIM * sizeof(bf16));
  bf16* w_proj = (bf16*)alloc((size_t)DIM * DIM * sizeof(bf16));
  bf16* w_fc1  = (bf16*)alloc((size_t)HID * DIM * sizeof(bf16));
  bf16* w_fc2  = (bf16*)alloc((size_t)DIM * HID * sizeof(bf16));
  bf16* a1     = (bf16*)alloc(MWIN * DIM * sizeof(bf16));          // LN1 out; reused as O
  char* big    = alloc(MWIN * 3 * DIM * sizeof(bf16));             // QKV; reused as x2+a2
  bf16* qkvb   = (bf16*)big;
  float* x2    = (float*)big;                                      // after attention done
  bf16* a2     = (bf16*)(big + NTOKENS * DIM * sizeof(float));
  bf16* hmid   = (bf16*)alloc(NTOKENS * HID * sizeof(bf16));
  bf16* o_attn = a1;  // a1 dead after QKV GEMM

  // Weight fp32 -> bf16
  cvt_f32_bf16_kernel<<<(3 * DIM * DIM + 255) / 256, 256, 0, stream>>>(qkv_w, w_qkv, 3 * DIM * DIM);
  cvt_f32_bf16_kernel<<<(DIM * DIM + 255) / 256, 256, 0, stream>>>(proj_w, w_proj, DIM * DIM);
  cvt_f32_bf16_kernel<<<(HID * DIM + 255) / 256, 256, 0, stream>>>(fc1_w, w_fc1, HID * DIM);
  cvt_f32_bf16_kernel<<<(DIM * HID + 255) / 256, 256, 0, stream>>>(fc2_w, w_fc2, DIM * HID);

  // LN1 + window partition (+ zero padded rows)
  ln1_window_kernel<<<BATCH * HW * 2, 256, 0, stream>>>(x, ln1_g, ln1_b, a1);
  zero_pad_kernel<<<(int)(((size_t)NWIN * 15 * DIM + 255) / 256), 256, 0, stream>>>(a1);

  // QKV GEMM: [131072,384] x [1152,384]^T -> bf16 [131072,1152]
  gemm_bf16_kernel<0><<<dim3(1152 / 128, (int)(MWIN / 128)), 256, 0, stream>>>(
      a1, w_qkv, qkv_b, DIM, 3 * DIM, qkvb, nullptr, nullptr);

  // Window attention (2048 windows x 12 heads)
  attn_kernel<<<dim3(NWIN, HEADS / 2), 64, 0, stream>>>(qkvb, o_attn);

  // Proj GEMM + un-window + residual -> x2 fp32 [100352,384]
  gemm_bf16_kernel<1><<<dim3(DIM / 128, (int)(MWIN / 128)), 256, 0, stream>>>(
      o_attn, w_proj, proj_b, DIM, DIM, nullptr, x2, x);

  // LN2
  ln2_kernel<<<(int)NTOKENS, DIM, 0, stream>>>(x2, ln2_g, ln2_b, a2);

  // FC1 + GELU -> bf16 [100352,1024]
  gemm_bf16_kernel<2><<<dim3(HID / 128, (int)(NTOKENS / 128)), 256, 0, stream>>>(
      a2, w_fc1, fc1_b, DIM, HID, hmid, nullptr, nullptr);

  // FC2 + residual -> fp32 NCHW output
  gemm_bf16_kernel<3><<<dim3(DIM / 128, (int)(NTOKENS / 128)), 256, 0, stream>>>(
      hmid, w_fc2, fc2_b, HID, DIM, nullptr, out, x2);
}
